// ScoreGraphReconstructor_35184372089222
// MI455X (gfx1250) — matrix operand order repaired and checked by reading, not compiled, not run on hardware
//
#include <hip/hip_runtime.h>
#include <math.h>

// Problem constants (from the reference)
#define N_NODES 20000
#define HEADS   4
#define CHAN    64
#define HIDF    256           // H*C
#define E_EDGES 160000
#define NCLS    5
#define EAUG    (E_EDGES + N_NODES)   // edges + self loops

typedef __attribute__((ext_vector_type(16))) _Float16 v16h;
typedef __attribute__((ext_vector_type(8)))  float    v8f;

union V16H { v16h v; uint4 q[2]; _Float16 h[16]; };

static __device__ __forceinline__ float lrelu02(float v) { return v > 0.f ? v : 0.2f * v; }
static __device__ __forceinline__ float eluf(float v)    { return v > 0.f ? v : __expf(v) - 1.f; }

static __device__ __forceinline__ void atomicMaxF(float* addr, float v) {
  // standard int-punned float max (init value must be -inf)
  if (v >= 0.f) atomicMax((int*)addr, __float_as_int(v));
  else          atomicMin((unsigned int*)addr, __float_as_uint(v));
}

// CDNA5 async memory->LDS copy (ASYNCcnt-tracked), per-lane 16 bytes.
static __device__ __forceinline__ void async_b128(unsigned lds_off, const void* gaddr) {
  asm volatile("global_load_async_to_lds_b128 %0, %1, off"
               :: "v"(lds_off), "v"((unsigned long long)(size_t)gaddr)
               : "memory");
}
static __device__ __forceinline__ void wait_async0() {
  asm volatile("s_wait_asynccnt 0x0" ::: "memory");
}
static __device__ __forceinline__ void wait_async6() {
  asm volatile("s_wait_asynccnt 0x6" ::: "memory");
}

// ---------------------------------------------------------------- utilities
__global__ void k_f32_to_f16(_Float16* __restrict__ dst, const float* __restrict__ src, int n) {
  int i = blockIdx.x * blockDim.x + threadIdx.x;
  if (i < n) dst[i] = (_Float16)src[i];
}

// transpose-convert: src is K x Nc (f32, row-major), dst is Nc x K (f16)
__global__ void k_f32_to_f16_T(_Float16* __restrict__ dst, const float* __restrict__ src,
                               int K, int Nc) {
  int idx = blockIdx.x * blockDim.x + threadIdx.x;
  if (idx >= K * Nc) return;
  int k = idx / Nc, n = idx - k * Nc;
  dst[(size_t)n * K + k] = (_Float16)src[idx];
}

__global__ void k_fill(float* __restrict__ p, float v, int n) {
  int i = blockIdx.x * blockDim.x + threadIdx.x;
  if (i < n) p[i] = v;
}

// ---------------------------------------------------------------- WMMA GEMM
// C[M,Nc] = A[M,K](f16,row-major) @ B (given TRANSPOSED: Bt[Nc,K] f16) + bias
// optional ELU ; f32 and/or f16 out.
// Block = 128 threads (4 waves). Block tile = 128x64, wave tile = 32x64,
// K-step 32. Double-buffered async global->LDS staging (ASYNCcnt pipelining).
__global__ __launch_bounds__(128)
void k_gemm_wmma(const _Float16* __restrict__ A, const _Float16* __restrict__ Bt,
                 const float* __restrict__ bias,
                 float* __restrict__ outF, _Float16* __restrict__ outH,
                 int M, int K, int Nc, int act)
{
  __shared__ _Float16 As[2][128][32];   // [buf][m][k]  (8 KB each)
  __shared__ _Float16 Bs[2][64][32];    // [buf][n][k]  (4 KB each)

  const int t    = threadIdx.x;
  const int wave = t >> 5;
  const int lane = t & 31;
  const int m0   = blockIdx.x * 128;
  const int n0   = blockIdx.y * 64;

  // staging: thread t stages A row t (32 halves = 4 async B128)
  //          and 16 halves of B row t/2 (2 async B128)
  int rr = m0 + t; if (rr > M - 1) rr = M - 1;        // clamp (guarded on store)
  const _Float16* gA = A + (size_t)rr * K;
  const int rb = t >> 1;
  const int cb = (t & 1) * 16;
  const _Float16* gB = Bt + (size_t)(n0 + rb) * K + cb;
  const unsigned ldsA[2] = { (unsigned)(size_t)&As[0][t][0],
                             (unsigned)(size_t)&As[1][t][0] };
  const unsigned ldsB[2] = { (unsigned)(size_t)&Bs[0][rb][cb],
                             (unsigned)(size_t)&Bs[1][rb][cb] };

  auto stage = [&](int buf, int k0) {
    const _Float16* a = gA + k0;
    async_b128(ldsA[buf],      a);
    async_b128(ldsA[buf] + 16, a + 8);
    async_b128(ldsA[buf] + 32, a + 16);
    async_b128(ldsA[buf] + 48, a + 24);
    const _Float16* b = gB + k0;
    async_b128(ldsB[buf],      b);
    async_b128(ldsB[buf] + 16, b + 8);
  };

  v8f acc[2][4] = {};    // 2 row-subtiles x 4 col-subtiles of 16x16

  stage(0, 0);
  int buf = 0;
  const int wr = wave << 5;            // wave's 32-row base within block tile
  const int kA = (lane >> 4) * 8;
  const int kB = (lane >> 4) * 16;
  const int l15 = lane & 15;

  for (int k0 = 0; k0 < K; k0 += 32) {
    if (k0 + 32 < K) { stage(buf ^ 1, k0 + 32); wait_async6(); }
    else             { wait_async0(); }
    __syncthreads();

    const _Float16 (*Asb)[32] = As[buf];
    const _Float16 (*Bsb)[32] = Bs[buf];

    // A fragments for two 16-row subtiles (ISA 16-bit A 16x32 layout:
    // lanes 0-15 K0..7/K16..23, lanes 16-31 K8..15/K24..31)
    V16H a0, a1;
    a0.q[0] = *(const uint4*)&Asb[wr + l15][kA];
    a0.q[1] = *(const uint4*)&Asb[wr + l15][16 + kA];
    a1.q[0] = *(const uint4*)&Asb[wr + 16 + l15][kA];
    a1.q[1] = *(const uint4*)&Asb[wr + 16 + l15][16 + kA];

#pragma unroll
    for (int nt = 0; nt < 4; ++nt) {
      V16H b;
      const int nrow = nt * 16 + l15;
      b.q[0] = *(const uint4*)&Bsb[nrow][kA];
      b.q[1] = *(const uint4*)&Bsb[nrow][16 + kA];
      acc[0][nt] = __builtin_amdgcn_wmma_f32_16x16x32_f16(
          false, a0.v, false, b.v, (short)0, acc[0][nt], false, false);
      acc[1][nt] = __builtin_amdgcn_wmma_f32_16x16x32_f16(
          false, a1.v, false, b.v, (short)0, acc[1][nt], false, false);
    }
    __syncthreads();
    buf ^= 1;
  }

  // ---- epilogue: VGPR i holds M=i (lanes 0-15) / M=i+8 (lanes 16-31)
  const int rbase = m0 + wr + ((lane >= 16) ? 8 : 0);
#pragma unroll
  for (int rt = 0; rt < 2; ++rt) {
#pragma unroll
    for (int nt = 0; nt < 4; ++nt) {
      const int col = n0 + nt * 16 + l15;
      const float bv = bias[col];
#pragma unroll
      for (int i = 0; i < 8; ++i) {
        const int row = rbase + rt * 16 + i;
        if (row < M) {
          float v = acc[rt][nt][i] + bv;
          if (act) v = eluf(v);
          if (outF) outF[(size_t)row * Nc + col] = v;
          if (outH) outH[(size_t)row * Nc + col] = (_Float16)v;
        }
      }
    }
  }
}

// ---------------------------------------------------------------- GAT edge kernels
// one thread per (edge, head): logit = sum_c lrelu(xl[src]+xr[dst]) * att[h]
__global__ void k_edge_logits(const float* __restrict__ xl, const float* __restrict__ xr,
                              const int* __restrict__ ei, const float* __restrict__ att,
                              float* __restrict__ logit, float* __restrict__ mx)
{
  int idx = blockIdx.x * blockDim.x + threadIdx.x;
  if (idx >= EAUG * HEADS) return;
  int e = idx >> 2, h = idx & 3;
  int s, d;
  if (e < E_EDGES) { s = ei[e]; d = ei[E_EDGES + e]; } else { s = d = e - E_EDGES; }
  const float4* xl4 = (const float4*)(xl + (size_t)s * HIDF + h * CHAN);
  const float4* xr4 = (const float4*)(xr + (size_t)d * HIDF + h * CHAN);
  const float4* at4 = (const float4*)(att + h * CHAN);
  float acc = 0.f;
#pragma unroll
  for (int q = 0; q < 16; ++q) {
    float4 a = xl4[q], b = xr4[q], w = at4[q];
    acc += lrelu02(a.x + b.x) * w.x + lrelu02(a.y + b.y) * w.y +
           lrelu02(a.z + b.z) * w.z + lrelu02(a.w + b.w) * w.w;
  }
  logit[idx] = acc;
  atomicMaxF(&mx[d * HEADS + h], acc);
}

__global__ void k_edge_exp(float* __restrict__ logit, const int* __restrict__ ei,
                           const float* __restrict__ mx, float* __restrict__ denom)
{
  int idx = blockIdx.x * blockDim.x + threadIdx.x;
  if (idx >= EAUG * HEADS) return;
  int e = idx >> 2, h = idx & 3;
  int d = (e < E_EDGES) ? ei[E_EDGES + e] : (e - E_EDGES);
  float ex = __expf(logit[idx] - mx[d * HEADS + h]);
  logit[idx] = ex;                      // reuse buffer for exp values
  atomicAdd(&denom[d * HEADS + h], ex);
}

// one thread per (edge, 4-channel group): hout[dst] += xl[src] * alpha
__global__ void k_edge_scatter(const float* __restrict__ xl, const int* __restrict__ ei,
                               const float* __restrict__ ex, const float* __restrict__ denom,
                               float* __restrict__ hout)
{
  int idx = blockIdx.x * blockDim.x + threadIdx.x;
  if (idx >= EAUG * (HIDF / 4)) return;
  int e = idx >> 6;
  int q = idx & 63;          // channel group; h = q/16
  int h = q >> 4;
  int s, d;
  if (e < E_EDGES) { s = ei[e]; d = ei[E_EDGES + e]; } else { s = d = e - E_EDGES; }
  float alpha = ex[(size_t)e * HEADS + h] / denom[d * HEADS + h];
  float4 xv = *(const float4*)(xl + (size_t)s * HIDF + q * 4);
  float* hp = hout + (size_t)d * HIDF + q * 4;
  atomicAdd(hp + 0, xv.x * alpha);
  atomicAdd(hp + 1, xv.y * alpha);
  atomicAdd(hp + 2, xv.z * alpha);
  atomicAdd(hp + 3, xv.w * alpha);
}

__global__ void k_finish(const float* __restrict__ hout, const float* __restrict__ bias,
                         float* __restrict__ hf, _Float16* __restrict__ h16, int n, int act)
{
  int idx = blockIdx.x * blockDim.x + threadIdx.x;
  if (idx >= n) return;
  int c = idx & (HIDF - 1);
  float v = hout[idx] + bias[c];
  if (act) v = eluf(v);
  hf[idx] = v;
  h16[idx] = (_Float16)v;
}

// ---------------------------------------------------------------- classifier
// ef[e] = concat(h[row[e]], h[col[e]]) as f16 ; one thread per (edge, 8 cols)
__global__ void k_gather_ef(const float* __restrict__ h, const int* __restrict__ ei,
                            _Float16* __restrict__ ef)
{
  int idx = blockIdx.x * blockDim.x + threadIdx.x;
  if (idx >= E_EDGES * 64) return;
  int e  = idx >> 6;
  int c0 = (idx & 63) * 8;               // 0..504
  int node = (c0 < HIDF) ? ei[e] : ei[E_EDGES + e];
  int cc = c0 & (HIDF - 1);
  const float4* hp = (const float4*)(h + (size_t)node * HIDF + cc);
  float4 a = hp[0], b = hp[1];
  _Float16 o[8] = { (_Float16)a.x, (_Float16)a.y, (_Float16)a.z, (_Float16)a.w,
                    (_Float16)b.x, (_Float16)b.y, (_Float16)b.z, (_Float16)b.w };
  *(uint4*)(ef + (size_t)e * (2 * HIDF) + c0) = *(uint4*)o;
}

// out[e] = z2[e] (E x 128, f16, ELU applied) @ Wc3 (128x5, f32) + bc3
__global__ void k_out_small(const _Float16* __restrict__ z2, const float* __restrict__ Wc3,
                            const float* __restrict__ bc3, float* __restrict__ out)
{
  int e = blockIdx.x * blockDim.x + threadIdx.x;
  if (e >= E_EDGES) return;
  float s0 = bc3[0], s1 = bc3[1], s2 = bc3[2], s3 = bc3[3], s4 = bc3[4];
  const _Float16* zp = z2 + (size_t)e * 128;
#pragma unroll 8
  for (int k = 0; k < 128; ++k) {
    float z = (float)zp[k];
    const float* wrow = Wc3 + k * NCLS;
    s0 += z * wrow[0]; s1 += z * wrow[1]; s2 += z * wrow[2];
    s3 += z * wrow[3]; s4 += z * wrow[4];
  }
  float* op = out + (size_t)e * NCLS;
  op[0] = s0; op[1] = s1; op[2] = s2; op[3] = s3; op[4] = s4;
}

// ---------------------------------------------------------------- host side
static inline int cdiv(int a, int b) { return (a + b - 1) / b; }

extern "C" void kernel_launch(void* const* d_in, const int* in_sizes, int n_in,
                              void* d_out, int out_size, void* d_ws, size_t ws_size,
                              hipStream_t stream) {
  (void)in_sizes; (void)n_in; (void)out_size; (void)ws_size;

  const float* x  = (const float*)d_in[0];
  const int*   ei = (const int*)d_in[1];
  const float* Wl[3]   = { (const float*)d_in[2],  (const float*)d_in[8],  (const float*)d_in[14] };
  const float* bl[3]   = { (const float*)d_in[3],  (const float*)d_in[9],  (const float*)d_in[15] };
  const float* Wr[3]   = { (const float*)d_in[4],  (const float*)d_in[10], (const float*)d_in[16] };
  const float* br[3]   = { (const float*)d_in[5],  (const float*)d_in[11], (const float*)d_in[17] };
  const float* attp[3] = { (const float*)d_in[6],  (const float*)d_in[12], (const float*)d_in[18] };
  const float* bias[3] = { (const float*)d_in[7],  (const float*)d_in[13], (const float*)d_in[19] };
  const float* Wc1 = (const float*)d_in[20]; const float* bc1 = (const float*)d_in[21];
  const float* Wc2 = (const float*)d_in[22]; const float* bc2 = (const float*)d_in[23];
  const float* Wc3 = (const float*)d_in[24]; const float* bc3 = (const float*)d_in[25];

  // ---- workspace carve-out (persistent region + union region)
  size_t off = 0;
  auto carve = [&](size_t bytes) -> char* {
    char* p = (char*)d_ws + off;
    off = (off + bytes + 255) & ~(size_t)255;
    return p;
  };
  _Float16* h16   = (_Float16*)carve((size_t)N_NODES * HIDF * 2);
  _Float16* wA16  = (_Float16*)carve((size_t)512 * 256 * 2);   // Wl^T / Wc1^T
  _Float16* wB16  = (_Float16*)carve((size_t)256 * 256 * 2);   // Wr^T / Wc2^T
  float*    mx    = (float*)carve((size_t)N_NODES * HEADS * 4);
  float*    denom = (float*)carve((size_t)N_NODES * HEADS * 4);
  float*    hf    = (float*)carve((size_t)N_NODES * HIDF * 4);
  char*     ub    = (char*)d_ws + off;              // union region
  // per-layer view
  float* xl    = (float*)ub;
  float* xr    = xl + (size_t)N_NODES * HIDF;
  float* logit = xr + (size_t)N_NODES * HIDF;       // EAUG*HEADS, reused for exp
  float* hout  = logit + (size_t)EAUG * HEADS;
  // classifier view (reuses the same region; layer buffers dead by then)
  _Float16* ef = (_Float16*)ub;
  _Float16* z1 = ef + (size_t)E_EDGES * 512;
  _Float16* z2 = z1 + (size_t)E_EDGES * 256;

  const int TB = 256;
  auto gemm = [&](const _Float16* A, const _Float16* Bt, const float* bv,
                  float* oF, _Float16* oH, int M, int K, int Nc, int act) {
    dim3 grid(cdiv(M, 128), Nc / 64);
    k_gemm_wmma<<<grid, dim3(128), 0, stream>>>(A, Bt, bv, oF, oH, M, K, Nc, act);
  };
  auto cvt = [&](_Float16* dst, const float* src, int n) {
    k_f32_to_f16<<<cdiv(n, TB), TB, 0, stream>>>(dst, src, n);
  };
  auto cvtT = [&](_Float16* dst, const float* src, int K, int Nc) {
    k_f32_to_f16_T<<<cdiv(K * Nc, TB), TB, 0, stream>>>(dst, src, K, Nc);
  };

  // layer-0 input
  cvt(h16, x, N_NODES * HIDF);

  for (int L = 0; L < 3; ++L) {
    cvtT(wA16, Wl[L], HIDF, HIDF);
    cvtT(wB16, Wr[L], HIDF, HIDF);
    gemm(h16, wA16, bl[L], xl, nullptr, N_NODES, HIDF, HIDF, 0);
    gemm(h16, wB16, br[L], xr, nullptr, N_NODES, HIDF, HIDF, 0);

    k_fill<<<cdiv(N_NODES * HEADS, TB), TB, 0, stream>>>(mx, -INFINITY, N_NODES * HEADS);
    k_fill<<<cdiv(N_NODES * HEADS, TB), TB, 0, stream>>>(denom, 0.f, N_NODES * HEADS);
    k_fill<<<cdiv(N_NODES * HIDF, TB), TB, 0, stream>>>(hout, 0.f, N_NODES * HIDF);

    k_edge_logits<<<cdiv(EAUG * HEADS, TB), TB, 0, stream>>>(xl, xr, ei, attp[L], logit, mx);
    k_edge_exp<<<cdiv(EAUG * HEADS, TB), TB, 0, stream>>>(logit, ei, mx, denom);
    k_edge_scatter<<<cdiv(EAUG * (HIDF / 4), TB), TB, 0, stream>>>(xl, ei, logit, denom, hout);
    k_finish<<<cdiv(N_NODES * HIDF, TB), TB, 0, stream>>>(hout, bias[L], hf, h16,
                                                          N_NODES * HIDF, (L < 2) ? 1 : 0);
  }

  // classifier
  k_gather_ef<<<cdiv(E_EDGES * 64, TB), TB, 0, stream>>>(hf, ei, ef);
  cvtT(wA16, Wc1, 2 * HIDF, HIDF);
  gemm(ef, wA16, bc1, nullptr, z1, E_EDGES, 2 * HIDF, HIDF, 1);
  cvtT(wB16, Wc2, HIDF, HIDF / 2);
  gemm(z1, wB16, bc2, nullptr, z2, E_EDGES, HIDF, HIDF / 2, 1);
  k_out_small<<<cdiv(E_EDGES, TB), TB, 0, stream>>>(z2, Wc3, bc3, (float*)d_out);
}
